// Actor_72679436583512
// MI455X (gfx1250) — compile-verified, matrix operand order checked by reading, not run on hardware
//
#include <hip/hip_runtime.h>
#include <hip/hip_bf16.h>
#include <math.h>

// ---------------------------------------------------------------------------
// Problem constants (from the reference)
// ---------------------------------------------------------------------------
#define BDIM  4096
#define REPR  39200
#define FEAT  256
#define HIDN  1024
#define GATEN 256
#define MHID  256
#define NEXP  32
#define TOPK  4
#define SDIM  24
#define ADIM  7

// Tiling
#define BM 64
#define BN 64
#define BK 32
#define LSTR 40    // LDS row stride (bf16 elems) for 32-wide tiles: 80B rows
#define ASTR 264   // LDS row stride (bf16 elems) for 256-wide TDM slab: 528B rows

typedef __attribute__((ext_vector_type(8)))  float        v8f;
typedef __attribute__((ext_vector_type(16))) __bf16       bf16x16;
typedef __attribute__((ext_vector_type(4)))  unsigned int u32x4;
typedef __attribute__((ext_vector_type(8)))  int          i32x8;
typedef __attribute__((ext_vector_type(4)))  int          i32x4;

union Frag { bf16x16 v; unsigned int u[8]; };

__device__ __forceinline__ unsigned short bf16r(float f) {
  unsigned int u = __float_as_uint(f);
  u += 0x7FFFu + ((u >> 16) & 1u);          // round-to-nearest-even
  return (unsigned short)(u >> 16);
}

// k-pair base offset for the bf16 A/B fragment layout (16-bit A 16x32):
// VGPR v<4 holds K = 2v,2v+1 (+8*khalf); v>=4 holds K = 16+2(v-4) (+8*khalf)
__device__ __forceinline__ int frag_kbase(int v, int khalf) {
  return ((v < 4) ? (v << 1) : (16 + ((v - 4) << 1))) + khalf * 8;
}

// ---------------------------------------------------------------------------
// gfx1250 async global->LDS copy (ASYNCcnt) + waits. LDS byte address = low
// 32 bits of the generic pointer (ISA: LDS aperture truncates addr[31:0]).
// ---------------------------------------------------------------------------
__device__ __forceinline__ void async_copy_b128(void* lds_dst, const void* gsrc) {
  unsigned int       loff  = (unsigned int)(unsigned long long)lds_dst;
  unsigned long long gaddr = (unsigned long long)gsrc;
  asm volatile("global_load_async_to_lds_b128 %0, %1, off"
               :: "v"(loff), "v"(gaddr) : "memory");
}

__device__ __forceinline__ void wait_async0() {
#if __has_builtin(__builtin_amdgcn_s_wait_asynccnt)
  __builtin_amdgcn_s_wait_asynccnt(0);
#else
  asm volatile("s_wait_asynccnt 0" ::: "memory");
#endif
}

__device__ __forceinline__ void wait_tensor0() {
#if __has_builtin(__builtin_amdgcn_s_wait_tensorcnt)
  __builtin_amdgcn_s_wait_tensorcnt(0);
#else
  asm volatile("s_wait_tensorcnt 0" ::: "memory");
#endif
}

// ---------------------------------------------------------------------------
// Load a 64-row x 256-col bf16 slab from global into LDS with 16B row pad
// (LDS rows of 528B). TDM path: one tensor_load_to_lds issued by wave 0,
// descriptor per CDNA5 ISA:
//   g0: count=1 | lds_addr | global_addr | type=2
//   g1: data_size=2B, pad_enable, pad_interval=6 (512B), pad_amount=3 (16B),
//       tensor_dim0=256, tensor_dim1=4096, tile_dim0=256, tile_dim1=64,
//       tensor_dim0_stride=256
// Toolchain here exposes the 6-arg builtin:
//   (u32x4, i32x8, i32x4, i32x4, i32x8, i32 cpol)
// Fallback: per-thread async b128 copies.
// ---------------------------------------------------------------------------
__device__ __forceinline__ void load_slab(unsigned short* lds_dst,
                                          const unsigned short* gsrc,
                                          int wave, int tid)
{
#if __has_builtin(__builtin_amdgcn_tensor_load_to_lds)
  if (wave == 0) {
    unsigned long long ga = (unsigned long long)gsrc;
    u32x4 g0;
    g0[0] = 1u;                                             // count=1 (valid)
    g0[1] = (unsigned int)(unsigned long long)lds_dst;      // lds_addr
    g0[2] = (unsigned int)(ga & 0xFFFFFFFFu);               // global_addr lo
    g0[3] = (unsigned int)((ga >> 32) & 0x01FFFFFFu) | (2u << 30); // hi | type=2
    i32x8 g1;
    g1[0] = (int)((1u << 16) | (1u << 20) | (6u << 22) | (3u << 25));
    g1[1] = (int)(256u << 16);        // tensor_dim0[15:0] = 256
    g1[2] = (int)(4096u << 16);       // tensor_dim0 hi=0 | tensor_dim1[15:0]=4096
    g1[3] = (int)(256u << 16);        // tensor_dim1 hi=0 | tile_dim0=256
    g1[4] = 64;                       // tile_dim1=64, tile_dim2=0
    g1[5] = 256;                      // tensor_dim0_stride lo
    g1[6] = 0;                        // stride0 hi | stride1 lo
    g1[7] = 0;
    i32x4 gz4; gz4[0] = 0; gz4[1] = 0; gz4[2] = 0; gz4[3] = 0;
    i32x8 gz8;
    gz8[0] = 0; gz8[1] = 0; gz8[2] = 0; gz8[3] = 0;
    gz8[4] = 0; gz8[5] = 0; gz8[6] = 0; gz8[7] = 0;
    __builtin_amdgcn_tensor_load_to_lds(g0, g1, gz4, gz4, gz8, 0);
  }
#else
  // 64 rows x 512B; 4 threads/row, 8 x 16B chunks each
  int r  = tid >> 2;
  int q  = tid & 3;
#pragma unroll
  for (int j = 0; j < 8; ++j) {
    int c = q * 64 + j * 8;                 // element offset in row
    async_copy_b128(&lds_dst[r * ASTR + c], &gsrc[(long)r * MHID + c]);
  }
#endif
  (void)wave; (void)tid;
}

__device__ __forceinline__ void wait_slab(int wave) {
#if __has_builtin(__builtin_amdgcn_tensor_load_to_lds)
  if (wave == 0) wait_tensor0();
#else
  wait_async0();
#endif
}

// ---------------------------------------------------------------------------
// Generic GEMM:  C[M×N] = act(A[M×K] @ W[K×N] + bias)
// fp32 inputs, bf16 WMMA compute, f32 accumulate, f32 or bf16 output.
// Batched over blockIdx.z via element strides (0 => shared operand).
// Block = 256 threads = 8 waves; block tile 64×64; wave tile 16×32.
// All uses have K % 4 == 0 and N % 4 == 0 => float4-granular bounds checks.
// ---------------------------------------------------------------------------
__global__ __launch_bounds__(256)
void gemm_bf16_kernel(const float* __restrict__ A,
                      const float* __restrict__ W,
                      const float* __restrict__ bias,
                      void* __restrict__ Cout,
                      int M, int N, int K,
                      long sA, long sW, long sBias, long sC,
                      int act, int out_bf16)
{
  __shared__ unsigned short As[BM * LSTR];   // [m][k] bf16
  __shared__ unsigned short Bs[BN * LSTR];   // [n][k] bf16 (W transposed)

  const long e = blockIdx.z;
  A += e * sA;
  W += e * sW;
  const float* biasb = bias + e * sBias;

  const int tid  = threadIdx.x;
  const int wave = tid >> 5;
  const int lane = tid & 31;
  const int rowBase = blockIdx.y * BM;
  const int colBase = blockIdx.x * BN;

  v8f acc0 = {}; v8f acc1 = {};

  for (int k0 = 0; k0 < K; k0 += BK) {
    // software prefetch of the next A tile
    if (k0 + BK < K) {
      int r = tid >> 3, c4 = (tid & 7) << 2;
      __builtin_prefetch(&A[(long)(rowBase + r) * K + (k0 + BK + c4)], 0, 3);
    }
    // ---- stage A tile: 64 rows x 32 k, fp32 -> bf16 (float4 granularity) ----
#pragma unroll
    for (int i = 0; i < 2; ++i) {
      int lin = tid + i * 256;              // 0..511
      int r   = lin >> 3;                   // 0..63
      int c4  = (lin & 7) << 2;             // 0..28
      int gk  = k0 + c4;
      float4 v = make_float4(0.f, 0.f, 0.f, 0.f);
      if (gk + 3 < K) v = *(const float4*)&A[(long)(rowBase + r) * K + gk];
      unsigned short* dst = &As[r * LSTR + c4];
      dst[0] = bf16r(v.x); dst[1] = bf16r(v.y);
      dst[2] = bf16r(v.z); dst[3] = bf16r(v.w);
    }
    // ---- stage W tile transposed: [k][n] global -> [n][k] LDS ----
#pragma unroll
    for (int i = 0; i < 2; ++i) {
      int lin = tid + i * 256;
      int kr  = lin >> 4;                   // 0..31
      int n4  = (lin & 15) << 2;            // 0..60
      int gk  = k0 + kr;
      int gn  = colBase + n4;
      float4 v = make_float4(0.f, 0.f, 0.f, 0.f);
      if (gk < K && gn + 3 < N) v = *(const float4*)&W[(long)gk * N + gn];
      Bs[(n4 + 0) * LSTR + kr] = bf16r(v.x);
      Bs[(n4 + 1) * LSTR + kr] = bf16r(v.y);
      Bs[(n4 + 2) * LSTR + kr] = bf16r(v.z);
      Bs[(n4 + 3) * LSTR + kr] = bf16r(v.w);
    }
    __syncthreads();

    // ---- build fragments and WMMA ----
    const int khalf = lane >> 4;
    const int mloc  = (wave & 3) * 16 + (lane & 15);
    const int nloc  = (wave >> 2) * 32 + (lane & 15);
    Frag a, b0, b1;
#pragma unroll
    for (int v = 0; v < 8; ++v) {
      int kb = frag_kbase(v, khalf);
      a.u[v]  = *(const unsigned int*)&As[mloc * LSTR + kb];
      b0.u[v] = *(const unsigned int*)&Bs[nloc * LSTR + kb];
      b1.u[v] = *(const unsigned int*)&Bs[(nloc + 16) * LSTR + kb];
    }
    acc0 = __builtin_amdgcn_wmma_f32_16x16x32_bf16(false, a.v, false, b0.v,
                                                   (short)0, acc0, false, false);
    acc1 = __builtin_amdgcn_wmma_f32_16x16x32_bf16(false, a.v, false, b1.v,
                                                   (short)0, acc1, false, false);
    __syncthreads();
  }

  // ---- epilogue: C layout => VGPR v, lanes 0-15: M=v; lanes 16-31: M=v+8 ----
  const int khalf = lane >> 4;
  const int mg0 = rowBase + (wave & 3) * 16 + 8 * khalf;
  const int n0  = colBase + (wave >> 2) * 32 + (lane & 15);
  const int n1  = n0 + 16;
  const float bv0 = (n0 < N) ? biasb[n0] : 0.f;
  const float bv1 = (n1 < N) ? biasb[n1] : 0.f;

  if (out_bf16) {
    unsigned short* O = (unsigned short*)Cout + e * sC;
#pragma unroll
    for (int v = 0; v < 8; ++v) {
      long m = mg0 + v;
      float c0 = acc0[v] + bv0, c1 = acc1[v] + bv1;
      if (act == 1) { c0 = fmaxf(c0, 0.f); c1 = fmaxf(c1, 0.f); }
      if (n0 < N) O[m * N + n0] = bf16r(c0);
      if (n1 < N) O[m * N + n1] = bf16r(c1);
    }
  } else {
    float* O = (float*)Cout + e * sC;
#pragma unroll
    for (int v = 0; v < 8; ++v) {
      long m = mg0 + v;
      float c0 = acc0[v] + bv0, c1 = acc1[v] + bv1;
      if (act == 1) { c0 = fmaxf(c0, 0.f); c1 = fmaxf(c1, 0.f); }
      if (n0 < N) O[m * N + n0] = c0;
      if (n1 < N) O[m * N + n1] = c1;
    }
  }
}

// ---------------------------------------------------------------------------
// MoE combine: x_moe[b,:] = sum_e comb[b,e] * (H1[e,b,:] @ e2w[e] + e2b[e])
// H1 (bf16) A-slabs are loaded by the Tensor Data Mover (double-buffered,
// DMA for expert e+1 overlaps WMMA on expert e). Gate weights folded into the
// per-expert epilogue so the [B,E,HID] intermediate never touches memory.
// ---------------------------------------------------------------------------
__global__ __launch_bounds__(256)
void moe_combine_kernel(const unsigned short* __restrict__ H1,  // [E][B][MHID] bf16
                        const float* __restrict__ e2w,          // [E][MHID][HIDN]
                        const float* __restrict__ e2b,          // [E][HIDN]
                        const float* __restrict__ comb,         // [B][NEXP]
                        float* __restrict__ xmoe)               // [B][HIDN]
{
  __shared__ unsigned short Aslab[2][BM * ASTR];   // 2 x 33 KB (TDM, padded rows)
  __shared__ unsigned short Bs[BN * LSTR];
  __shared__ float combS[BM * NEXP];               // 8 KB

  const int tid  = threadIdx.x;
  const int wave = tid >> 5;
  const int lane = tid & 31;
  const int rowBase = blockIdx.y * BM;
  const int colBase = blockIdx.x * BN;

  // stage the gate-weight tile once: 64x32 f32 = 8KB = 512 float4 (async)
#pragma unroll
  for (int i = 0; i < 2; ++i) {
    int lin = tid + i * 256;               // 0..511
    int r = lin >> 3, c4 = (lin & 7) << 2;
    async_copy_b128(&combS[r * NEXP + c4],
                    &comb[(long)(rowBase + r) * NEXP + c4]);
  }
  wait_async0();

  // first expert slab via TDM
  load_slab(&Aslab[0][0], H1 + (long)rowBase * MHID, wave, tid);
  wait_slab(wave);
  __syncthreads();

  const int khalf = lane >> 4;
  const int mloc  = (wave & 3) * 16 + (lane & 15);
  const int nloc  = (wave >> 2) * 32 + (lane & 15);
  const int n0 = colBase + nloc;
  const int n1 = n0 + 16;
  const int mrow = (wave & 3) * 16 + 8 * khalf;

  v8f tot0 = {}; v8f tot1 = {};

  for (int ex = 0; ex < NEXP; ++ex) {
    const unsigned short* Acur = &Aslab[ex & 1][0];
    // kick off DMA of the next expert's slab while we compute this one
    if (ex + 1 < NEXP)
      load_slab(&Aslab[(ex + 1) & 1][0],
                H1 + ((long)(ex + 1) * BDIM + rowBase) * MHID, wave, tid);

    const float* We = e2w + (long)ex * MHID * HIDN;
    v8f c0 = {}; v8f c1 = {};

    for (int k0 = 0; k0 < MHID; k0 += BK) {
      // W tile transposed fp32 -> bf16 (VALU convert path)
#pragma unroll
      for (int i = 0; i < 2; ++i) {
        int lin = tid + i * 256;
        int kr = lin >> 4, n4 = (lin & 15) << 2;
        const float* src = &We[(long)(k0 + kr) * HIDN + colBase + n4];
        float4 v = *(const float4*)src;
        Bs[(n4 + 0) * LSTR + kr] = bf16r(v.x);
        Bs[(n4 + 1) * LSTR + kr] = bf16r(v.y);
        Bs[(n4 + 2) * LSTR + kr] = bf16r(v.z);
        Bs[(n4 + 3) * LSTR + kr] = bf16r(v.w);
      }
      __syncthreads();

      Frag a, b0, b1;
#pragma unroll
      for (int v = 0; v < 8; ++v) {
        int kb = frag_kbase(v, khalf);
        a.u[v]  = *(const unsigned int*)&Acur[mloc * ASTR + k0 + kb];
        b0.u[v] = *(const unsigned int*)&Bs[nloc * LSTR + kb];
        b1.u[v] = *(const unsigned int*)&Bs[(nloc + 16) * LSTR + kb];
      }
      c0 = __builtin_amdgcn_wmma_f32_16x16x32_bf16(false, a.v, false, b0.v,
                                                   (short)0, c0, false, false);
      c1 = __builtin_amdgcn_wmma_f32_16x16x32_bf16(false, a.v, false, b1.v,
                                                   (short)0, c1, false, false);
      __syncthreads();
    }

    // fold gate weight + expert bias into the running accumulator
    const float bv0 = e2b[ex * HIDN + n0];
    const float bv1 = e2b[ex * HIDN + n1];
#pragma unroll
    for (int v = 0; v < 8; ++v) {
      float w = combS[(mrow + v) * NEXP + ex];
      tot0[v] += w * (c0[v] + bv0);
      tot1[v] += w * (c1[v] + bv1);
    }

    // next slab must be fully in LDS before anyone reads it
    wait_slab(wave);
    __syncthreads();
  }

#pragma unroll
  for (int v = 0; v < 8; ++v) {
    long m = rowBase + mrow + v;
    xmoe[m * HIDN + n0] = tot0[v];
    xmoe[m * HIDN + n1] = tot1[v];
  }
}

// ---------------------------------------------------------------------------
// LayerNorm + tanh + residual add:  H = tanh(LN(T)*g+b) + S   (FEAT == 256)
// ---------------------------------------------------------------------------
__global__ __launch_bounds__(256)
void ln_tanh_add_kernel(const float* __restrict__ T, const float* __restrict__ S,
                        const float* __restrict__ g, const float* __restrict__ b,
                        float* __restrict__ H)
{
  const int row = blockIdx.x, t = threadIdx.x;
  __shared__ float red[256];
  float x = T[(long)row * FEAT + t];
  red[t] = x; __syncthreads();
  for (int s = 128; s > 0; s >>= 1) { if (t < s) red[t] += red[t + s]; __syncthreads(); }
  float mu = red[0] * (1.f / FEAT);
  __syncthreads();
  float d = x - mu;
  red[t] = d * d; __syncthreads();
  for (int s = 128; s > 0; s >>= 1) { if (t < s) red[t] += red[t + s]; __syncthreads(); }
  float var = red[0] * (1.f / FEAT);
  float y = d * rsqrtf(var + 1e-5f) * g[t] + b[t];
  H[(long)row * FEAT + t] = tanhf(y) + S[(long)row * FEAT + t];
}

// ---------------------------------------------------------------------------
// Gate: top-4-of-32, softmax over top-k -> comb; atomic sums for aux loss.
// One wave32 per token: lane == expert.
// ---------------------------------------------------------------------------
__global__ __launch_bounds__(32)
void gate_kernel(const float* __restrict__ logits, float* __restrict__ comb,
                 float* __restrict__ impSum, float* __restrict__ loadSum)
{
  const int b = blockIdx.x, e = threadIdx.x;
  float v = logits[(long)b * NEXP + e];

  // full softmax for importance
  float m = v;
  for (int o = 16; o > 0; o >>= 1) m = fmaxf(m, __shfl_xor(m, o, 32));
  float ex = __expf(v - m);
  float s = ex;
  for (int o = 16; o > 0; o >>= 1) s += __shfl_xor(s, o, 32);
  float prob = ex / s;

  // iterative top-4 (wave argmax)
  float val = v; int sel = 0; float topv[TOPK];
#pragma unroll
  for (int i = 0; i < TOPK; ++i) {
    float mv = val; int mi = e;
    for (int o = 16; o > 0; o >>= 1) {
      float ov = __shfl_xor(mv, o, 32);
      int   oi = __shfl_xor(mi, o, 32);
      if (ov > mv || (ov == mv && oi < mi)) { mv = ov; mi = oi; }
    }
    topv[i] = mv;
    if (e == mi) { sel = i + 1; val = -INFINITY; }
  }
  float tm = topv[0];
  float ts = 0.f;
#pragma unroll
  for (int i = 0; i < TOPK; ++i) ts += __expf(topv[i] - tm);
  float w = sel ? __expf(topv[sel - 1] - tm) / ts : 0.f;

  comb[(long)b * NEXP + e] = w;
  atomicAdd(&impSum[e], prob);
  atomicAdd(&loadSum[e], sel ? 1.f : 0.f);
}

__global__ void zero64_kernel(float* p) { p[threadIdx.x] = 0.f; }

__global__ __launch_bounds__(32)
void aux_kernel(const float* __restrict__ impSum, const float* __restrict__ loadSum,
                float* __restrict__ out)
{
  const int e = threadIdx.x;
  float v = (impSum[e] * (1.f / BDIM)) * (loadSum[e] * (1.f / BDIM));
  for (int o = 16; o > 0; o >>= 1) v += __shfl_xor(v, o, 32);
  if (e == 0) out[0] = (float)NEXP * v;
}

// ---------------------------------------------------------------------------
// Policy head: mu = tanh(relu(x_moe) @ p2_w + p2_b), N = 7
// ---------------------------------------------------------------------------
__global__ __launch_bounds__(256)
void policy2_kernel(const float* __restrict__ xmoe, const float* __restrict__ w,
                    const float* __restrict__ b, float* __restrict__ mu)
{
  const int bt = blockIdx.x, t = threadIdx.x;
  float acc[ADIM];
#pragma unroll
  for (int o = 0; o < ADIM; ++o) acc[o] = 0.f;
  for (int k = t; k < HIDN; k += 256) {
    float x = fmaxf(xmoe[(long)bt * HIDN + k], 0.f);
    const float* wr = &w[(long)k * ADIM];
#pragma unroll
    for (int o = 0; o < ADIM; ++o) acc[o] += x * wr[o];
  }
  __shared__ float red[256];
#pragma unroll
  for (int o = 0; o < ADIM; ++o) {
    red[t] = acc[o]; __syncthreads();
    for (int s = 128; s > 0; s >>= 1) { if (t < s) red[t] += red[t + s]; __syncthreads(); }
    if (t == 0) mu[(long)bt * ADIM + o] = tanhf(red[0] + b[o]);
    __syncthreads();
  }
}

// ---------------------------------------------------------------------------
// Host-side launcher
// ---------------------------------------------------------------------------
extern "C" void kernel_launch(void* const* d_in, const int* in_sizes, int n_in,
                              void* d_out, int out_size, void* d_ws, size_t ws_size,
                              hipStream_t stream)
{
  const float* obs       = (const float*)d_in[0];
  // d_in[1] = std (unused: mu is the TruncatedNormal mean)
  const float* obs_sens  = (const float*)d_in[2];
  const float* trunk_w   = (const float*)d_in[3];
  const float* trunk_b   = (const float*)d_in[4];
  const float* ln_g      = (const float*)d_in[5];
  const float* ln_b      = (const float*)d_in[6];
  const float* se1_w     = (const float*)d_in[7];
  const float* se1_b     = (const float*)d_in[8];
  const float* se2_w     = (const float*)d_in[9];
  const float* se2_b     = (const float*)d_in[10];
  const float* fu1_w     = (const float*)d_in[11];
  const float* fu1_b     = (const float*)d_in[12];
  const float* fu2_w     = (const float*)d_in[13];
  const float* fu2_b     = (const float*)d_in[14];
  const float* p1_w      = (const float*)d_in[15];
  const float* p1_b      = (const float*)d_in[16];
  const float* g1_w      = (const float*)d_in[17];
  const float* g1_b      = (const float*)d_in[18];
  const float* g2_w      = (const float*)d_in[19];
  const float* g2_b      = (const float*)d_in[20];
  const float* e1_w      = (const float*)d_in[21];
  const float* e1_b      = (const float*)d_in[22];
  const float* e2_w      = (const float*)d_in[23];
  const float* e2_b      = (const float*)d_in[24];
  const float* p2_w      = (const float*)d_in[25];
  const float* p2_b      = (const float*)d_in[26];

  // workspace carve-up (256B aligned)
  char* p = (char*)d_ws;
  auto carve = [&](size_t bytes) -> void* {
    void* r = (void*)p;
    p += (bytes + 255) & ~(size_t)255;
    return r;
  };
  float*          T    = (float*)carve((size_t)BDIM * FEAT * 4);   // trunk pre-LN
  float*          Sbuf = (float*)carve((size_t)BDIM * FEAT * 4);   // state-encoder out
  float*          H    = (float*)carve((size_t)BDIM * FEAT * 4);   // fused features
  float*          H2   = (float*)carve((size_t)BDIM * FEAT * 4);
  float*          X    = (float*)carve((size_t)BDIM * HIDN * 4);   // S1 / F1 / policy1
  float*          G    = (float*)carve((size_t)BDIM * GATEN * 4);
  float*          LOG  = (float*)carve((size_t)BDIM * NEXP * 4);
  float*          COMB = (float*)carve((size_t)BDIM * NEXP * 4);
  float*          XMOE = (float*)carve((size_t)BDIM * HIDN * 4);
  unsigned short* H1   = (unsigned short*)carve((size_t)NEXP * BDIM * MHID * 2); // bf16
  float*          STAT = (float*)carve(64 * 4);                    // impSum[32] ++ loadSum[32]

  dim3 blk(256);
  auto gemm_grid = [](int M, int N, int z) {
    return dim3((unsigned)((N + BN - 1) / BN), (unsigned)((M + BM - 1) / BM), (unsigned)z);
  };

  zero64_kernel<<<1, 64, 0, stream>>>(STAT);

  // trunk: T = obs @ trunk_w + trunk_b
  gemm_bf16_kernel<<<gemm_grid(BDIM, FEAT, 1), blk, 0, stream>>>(
      obs, trunk_w, trunk_b, T, BDIM, FEAT, REPR, 0, 0, 0, 0, /*act*/0, /*bf16*/0);
  // S1 = relu(obs_sensor @ se1_w + se1_b)
  gemm_bf16_kernel<<<gemm_grid(BDIM, HIDN, 1), blk, 0, stream>>>(
      obs_sens, se1_w, se1_b, X, BDIM, HIDN, SDIM, 0, 0, 0, 0, 1, 0);
  // S = S1 @ se2_w + se2_b
  gemm_bf16_kernel<<<gemm_grid(BDIM, FEAT, 1), blk, 0, stream>>>(
      X, se2_w, se2_b, Sbuf, BDIM, FEAT, HIDN, 0, 0, 0, 0, 0, 0);
  // H = tanh(LN(T)) + S
  ln_tanh_add_kernel<<<BDIM, 256, 0, stream>>>(T, Sbuf, ln_g, ln_b, H);
  // F1 = relu(H @ fu1_w + fu1_b)
  gemm_bf16_kernel<<<gemm_grid(BDIM, HIDN, 1), blk, 0, stream>>>(
      H, fu1_w, fu1_b, X, BDIM, HIDN, FEAT, 0, 0, 0, 0, 1, 0);
  // H2 = F1 @ fu2_w + fu2_b
  gemm_bf16_kernel<<<gemm_grid(BDIM, FEAT, 1), blk, 0, stream>>>(
      X, fu2_w, fu2_b, H2, BDIM, FEAT, HIDN, 0, 0, 0, 0, 0, 0);
  // X = relu(H2 @ p1_w + p1_b)
  gemm_bf16_kernel<<<gemm_grid(BDIM, HIDN, 1), blk, 0, stream>>>(
      H2, p1_w, p1_b, X, BDIM, HIDN, FEAT, 0, 0, 0, 0, 1, 0);
  // G = relu(X @ g1_w + g1_b)
  gemm_bf16_kernel<<<gemm_grid(BDIM, GATEN, 1), blk, 0, stream>>>(
      X, g1_w, g1_b, G, BDIM, GATEN, HIDN, 0, 0, 0, 0, 1, 0);
  // LOG = G @ g2_w + g2_b
  gemm_bf16_kernel<<<gemm_grid(BDIM, NEXP, 1), blk, 0, stream>>>(
      G, g2_w, g2_b, LOG, BDIM, NEXP, GATEN, 0, 0, 0, 0, 0, 0);
  // gating (comb + aux statistics)
  gate_kernel<<<BDIM, 32, 0, stream>>>(LOG, COMB, STAT, STAT + 32);
  // H1[e] = relu(X @ e1_w[e] + e1_b[e])  -> bf16, batched over experts
  gemm_bf16_kernel<<<gemm_grid(BDIM, MHID, NEXP), blk, 0, stream>>>(
      X, e1_w, e1_b, H1, BDIM, MHID, HIDN,
      /*sA*/0, /*sW*/(long)HIDN * MHID, /*sBias*/(long)MHID,
      /*sC*/(long)BDIM * MHID, 1, /*bf16*/1);
  // x_moe = sum_e comb[:,e] * (H1[e] @ e2_w[e] + e2_b[e])   (TDM-fed)
  moe_combine_kernel<<<gemm_grid(BDIM, HIDN, 1), blk, 0, stream>>>(
      H1, e2_w, e2_b, COMB, XMOE);
  // mu = tanh(relu(x_moe) @ p2_w + p2_b)
  policy2_kernel<<<BDIM, 256, 0, stream>>>(XMOE, p2_w, p2_b, (float*)d_out);
  // aux scalar appended after mu
  aux_kernel<<<1, 32, 0, stream>>>(STAT, STAT + 32, (float*)d_out + (size_t)BDIM * ADIM);

  (void)in_sizes; (void)n_in; (void)out_size; (void)ws_size;
}
